// tri_att_81200651698747
// MI455X (gfx1250) — matrix operand order repaired and checked by reading, not compiled, not run on hardware
//
#include <hip/hip_runtime.h>
#include <hip/hip_bf16.h>

// ---------------------------------------------------------------------------
// tri-attention structure map for x:(32,512,56,56) fp32
//   out[n,s] = sum_d w[n,d] * x[n,d,s],  w[n,d] = sum_c softmax_d( R[n,c,d]/Z_c )
//   R[n,c,d] = sum_s exp(7 x[n,c,s] - m_c) * x[n,d,s]
// GEMM runs on v_wmma_f32_16x16x32_bf16 (f32 accumulate).
// ---------------------------------------------------------------------------

typedef __attribute__((ext_vector_type(16))) __bf16 v16bf;
typedef __attribute__((ext_vector_type(8)))  __bf16 v8bf;
typedef __attribute__((ext_vector_type(8)))  float  v8f;
typedef __attribute__((ext_vector_type(4)))  float  v4f;

#define N_BATCH 32
#define C_DIM   512
#define S_DIM   3136   // 56*56, exactly 98 * 32
#define TM 64
#define TN 64
#define TK 32
#define LDSP 40        // bf16 elems per LDS row: 80 B pitch (16B aligned, conflict-free)

// ---------------- kernel 1: per-row online max / sum-exp of 7*x -------------
__global__ __launch_bounds__(256)
void row_stats_kernel(const float* __restrict__ x,
                      float* __restrict__ rowmax, float* __restrict__ rowsum) {
    __shared__ float sm[256], ss[256];
    const int row = blockIdx.x;                     // n*512 + c
    const float* xr = x + (size_t)row * S_DIM;
    float m = -1e30f, s = 0.f;
    for (int i = threadIdx.x; i < S_DIM; i += 256) {
        float v = 7.f * xr[i];
        if (v > m) { s = s * __expf(m - v) + 1.f; m = v; }
        else       { s += __expf(v - m); }
    }
    sm[threadIdx.x] = m; ss[threadIdx.x] = s;
    __syncthreads();
    for (int off = 128; off > 0; off >>= 1) {
        if (threadIdx.x < off) {
            float m2 = sm[threadIdx.x + off], s2 = ss[threadIdx.x + off];
            float m1 = sm[threadIdx.x],       s1 = ss[threadIdx.x];
            float M = fmaxf(m1, m2);
            ss[threadIdx.x] = s1 * __expf(m1 - M) + s2 * __expf(m2 - M);
            sm[threadIdx.x] = M;
        }
        __syncthreads();
    }
    if (threadIdx.x == 0) { rowmax[row] = sm[0]; rowsum[row] = ss[0]; }
}

// ---------------- kernel 2: WMMA GEMM  R[c,d] = sum_s e[c,s] * f[d,s] -------
__global__ __launch_bounds__(256)
void gemm_bilinear_kernel(const float* __restrict__ x,
                          const float* __restrict__ rowmax,
                          float* __restrict__ R) {
    __shared__ alignas(16) __bf16 As[TM * LDSP];
    __shared__ alignas(16) __bf16 Bs[TN * LDSP];

    const int tid  = threadIdx.x;
    const int lane = tid & 31;
    const int wave = tid >> 5;
    const int n    = blockIdx.z;
    const int c0   = blockIdx.y * TM;   // softmax-weighted rows (M)
    const int d0   = blockIdx.x * TN;   // key rows (N)

    const float* xb = x + (size_t)n * C_DIM * S_DIM;

    // cooperative loader: 64 rows x 32 cols fp32, 8 floats/thread
    const int lrow = tid >> 2;          // 0..63
    const int lcol = (tid & 3) * 8;     // 0,8,16,24
    const float mrow = rowmax[n * C_DIM + c0 + lrow];

    // wave fragment mapping: 16x16 tiles; each wave owns 16(M) x 32(N)
    const int mrow0 = (wave & 3) * 16;
    const int ncol0 = (wave >> 2) * 32;
    const int fr    = lane & 15;
    const int half  = lane >> 4;

    v8f acc0 = {}; v8f acc1 = {};

    for (int kc = 0; kc < S_DIM; kc += TK) {
        __syncthreads();
        { // A tile: exp(7x - m) -> bf16
            const float* src = xb + (size_t)(c0 + lrow) * S_DIM + kc + lcol;
            v4f a0 = *(const v4f*)src;
            v4f a1 = *(const v4f*)(src + 4);
            v8bf st;
            #pragma unroll
            for (int i = 0; i < 4; ++i) {
                st[i]     = (__bf16)__expf(7.f * a0[i] - mrow);
                st[i + 4] = (__bf16)__expf(7.f * a1[i] - mrow);
            }
            *(v8bf*)&As[lrow * LDSP + lcol] = st;
        }
        { // B tile: raw x -> bf16
            const float* src = xb + (size_t)(d0 + lrow) * S_DIM + kc + lcol;
            v4f b0 = *(const v4f*)src;
            v4f b1 = *(const v4f*)(src + 4);
            v8bf st;
            #pragma unroll
            for (int i = 0; i < 4; ++i) {
                st[i]     = (__bf16)b0[i];
                st[i + 4] = (__bf16)b1[i];
            }
            *(v8bf*)&Bs[lrow * LDSP + lcol] = st;
        }
        if (kc + TK < S_DIM) { // prefetch next K chunk (global_prefetch_b8)
            __builtin_prefetch(xb + (size_t)(c0 + lrow) * S_DIM + kc + TK + lcol, 0, 3);
            __builtin_prefetch(xb + (size_t)(d0 + lrow) * S_DIM + kc + TK + lcol, 0, 3);
        }
        __syncthreads();

        // Fragments in contiguous k order (same for A and B):
        //  A 16x32 bf16: lane (m=fr, half): elems 0..15 -> K=16*half+e
        //  B 32x16 bf16: lane (n=fr, half): elems 0..15 -> K=16*half+e
        union V16 { v16bf v; v8bf h[2]; };
        V16 a, b0f, b1f;
        a.h[0]   = *(const v8bf*)&As[(mrow0 + fr) * LDSP + 16 * half];
        a.h[1]   = *(const v8bf*)&As[(mrow0 + fr) * LDSP + 16 * half + 8];
        b0f.h[0] = *(const v8bf*)&Bs[(ncol0 + fr) * LDSP + 16 * half];
        b0f.h[1] = *(const v8bf*)&Bs[(ncol0 + fr) * LDSP + 16 * half + 8];
        b1f.h[0] = *(const v8bf*)&Bs[(ncol0 + 16 + fr) * LDSP + 16 * half];
        b1f.h[1] = *(const v8bf*)&Bs[(ncol0 + 16 + fr) * LDSP + 16 * half + 8];

        acc0 = __builtin_amdgcn_wmma_f32_16x16x32_bf16(false, a.v, false, b0f.v,
                                                       (short)0, acc0, false, false);
        acc1 = __builtin_amdgcn_wmma_f32_16x16x32_bf16(false, a.v, false, b1f.v,
                                                       (short)0, acc1, false, false);
    }

    // C/D layout: VGPR r -> row mrow0 + 8*half + r, col = fr (lanes%16)
    float* Rn = R + ((size_t)n * C_DIM + c0) * C_DIM + d0;
    #pragma unroll
    for (int r = 0; r < 8; ++r) {
        int row = mrow0 + 8 * half + r;
        Rn[(size_t)row * C_DIM + ncol0 + fr]      = acc0[r];
        Rn[(size_t)row * C_DIM + ncol0 + 16 + fr] = acc1[r];
    }
}

// ---------------- kernel 3: softmax over d of R/Z, accumulate column sums ---
__global__ __launch_bounds__(256)
void softmax_colsum_kernel(const float* __restrict__ R,
                           const float* __restrict__ rowsum,
                           float* __restrict__ w) {
    __shared__ float wacc[C_DIM];
    __shared__ float red[256];
    const int n  = blockIdx.y;
    const int c0 = blockIdx.x * 16;
    const int t  = threadIdx.x;
    wacc[t] = 0.f; wacc[t + 256] = 0.f;
    __syncthreads();
    for (int i = 0; i < 16; ++i) {
        const int c = c0 + i;
        const float invZ = 1.f / rowsum[n * C_DIM + c];
        const float* Rr = R + ((size_t)n * C_DIM + c) * C_DIM;
        float v0 = Rr[t] * invZ;
        float v1 = Rr[t + 256] * invZ;
        red[t] = fmaxf(v0, v1);
        __syncthreads();
        for (int off = 128; off > 0; off >>= 1) {
            if (t < off) red[t] = fmaxf(red[t], red[t + off]);
            __syncthreads();
        }
        float m = red[0];
        __syncthreads();
        float e0 = __expf(v0 - m), e1 = __expf(v1 - m);
        red[t] = e0 + e1;
        __syncthreads();
        for (int off = 128; off > 0; off >>= 1) {
            if (t < off) red[t] += red[t + off];
            __syncthreads();
        }
        float invS = 1.f / red[0];
        __syncthreads();
        wacc[t]       += e0 * invS;
        wacc[t + 256] += e1 * invS;
    }
    atomicAdd(&w[n * C_DIM + t],       wacc[t]);
    atomicAdd(&w[n * C_DIM + t + 256], wacc[t + 256]);
}

// ---------------- kernel 4: out[n,s] = sum_d w[n,d] * x[n,d,s] --------------
__global__ __launch_bounds__(256)
void weighted_sum_kernel(const float* __restrict__ x,
                         const float* __restrict__ w,
                         float* __restrict__ out) {
    __shared__ float wl[C_DIM];
    const int n = blockIdx.y;
    const int t = threadIdx.x;
    wl[t]       = w[n * C_DIM + t];
    wl[t + 256] = w[n * C_DIM + t + 256];
    __syncthreads();
    const int s = blockIdx.x * 256 + t;
    if (s < S_DIM) {
        const float* xn = x + (size_t)n * C_DIM * S_DIM + s;
        float acc = 0.f;
        #pragma unroll 4
        for (int d = 0; d < C_DIM; ++d) acc += wl[d] * xn[(size_t)d * S_DIM];
        out[n * S_DIM + s] = acc;
    }
}

__global__ __launch_bounds__(256)
void zero_kernel(float* __restrict__ p, int nelem) {
    int i = blockIdx.x * 256 + threadIdx.x;
    if (i < nelem) p[i] = 0.f;
}

// ---------------------------------------------------------------------------
extern "C" void kernel_launch(void* const* d_in, const int* in_sizes, int n_in,
                              void* d_out, int out_size, void* d_ws, size_t ws_size,
                              hipStream_t stream) {
    const float* x = (const float*)d_in[0];
    float* out = (float*)d_out;
    float* ws  = (float*)d_ws;

    // workspace layout (floats)
    float* rowmax = ws;                       // 32*512
    float* rowsum = ws + N_BATCH * C_DIM;     // 32*512
    float* w      = ws + 2 * N_BATCH * C_DIM; // 32*512
    float* R      = ws + 3 * N_BATCH * C_DIM; // 32*512*512 (33.5 MB)

    row_stats_kernel<<<dim3(N_BATCH * C_DIM), 256, 0, stream>>>(x, rowmax, rowsum);
    zero_kernel<<<dim3((N_BATCH * C_DIM + 255) / 256), 256, 0, stream>>>(w, N_BATCH * C_DIM);
    gemm_bilinear_kernel<<<dim3(C_DIM / TN, C_DIM / TM, N_BATCH), 256, 0, stream>>>(x, rowmax, R);
    softmax_colsum_kernel<<<dim3(C_DIM / 16, N_BATCH), 256, 0, stream>>>(R, rowsum, w);
    weighted_sum_kernel<<<dim3((S_DIM + 255) / 256, N_BATCH), 256, 0, stream>>>(x, w, out);
}
